// OneByOneMergeConv3D_20538533609778
// MI455X (gfx1250) — matrix operand order repaired and checked by reading, not compiled, not run on hardware
//
#include <hip/hip_runtime.h>
#include <math.h>

// ---------------------------------------------------------------------------
// Problem: B=8, H=128, W=128, C=64, C3=192, SE_R=16 (hidden=12)
//   x = interleave(t1,t2,t3) [B,H,W,192]
//   s = sigmoid(relu(mean_hw(x) @ w_se1) @ w_se2)        [B,192]
//   out = relu((x * s) @ w_conv + b_conv)                 [B,H,W,64]
// Strategy: fold s (and the channel interleave) into per-batch bf16 weights
//   Wt[b][f][k'] = s[b, j(k')] * w_conv[j(k'), f],  k' = t*64+c, j = 3c+t
// then main pass is a memory-bound per-pixel 192->64 bf16 WMMA matmul.
// Weight slab staged into LDS via the Tensor Data Mover (TENSOR_LOAD_TO_LDS).
// ---------------------------------------------------------------------------

#define BB 8
#define HH 128
#define WW 128
#define CC 64
#define C3 192
#define HID 12
#define PIX_PER_B (HH * WW)            // 16384
#define NPIX (BB * PIX_PER_B)          // 131072
#define TILE_PIX 128                   // pixels per workgroup in main kernel
#define WT_ELEMS (BB * CC * C3)        // 98304 bf16
#define WT_PER_B (CC * C3)             // 12288 bf16 = 24 KB
#define SUMS_ELEMS (BB * C3)           // 1536 f32
#define WT_BYTE_OFF 6144               // sums f32 region size in bytes

typedef __attribute__((ext_vector_type(16))) __bf16 v16bf;
typedef __attribute__((ext_vector_type(8)))  __bf16 v8bf;
typedef __attribute__((ext_vector_type(8)))  float  v8f;
typedef __attribute__((ext_vector_type(4)))  unsigned int u32x4;
typedef __attribute__((ext_vector_type(8)))  int        i32x8;
typedef __attribute__((ext_vector_type(4)))  int        i32x4;

union V16U { v16bf v; v8bf h[2]; };

__device__ __forceinline__ unsigned short f2bf(float f) {
    unsigned int u = __float_as_uint(f);
    unsigned int r = u + 0x7FFFu + ((u >> 16) & 1u);   // round-to-nearest-even
    return (unsigned short)(r >> 16);
}

struct __align__(8) US4 { unsigned short a, b, c, d; };

// ---------------------------------------------------------------------------
// Kernel 0: zero the f32 accumulator region of the workspace (it is poisoned)
// ---------------------------------------------------------------------------
__global__ __launch_bounds__(256) void k0_zero(float* __restrict__ sums) {
    int i = blockIdx.x * 256 + threadIdx.x;
    if (i < SUMS_ELEMS) sums[i] = 0.0f;
}

// ---------------------------------------------------------------------------
// Kernel 1: global-average-pool partial sums.
// grid = 8 batches * 3 tensors * 128 chunks = 3072 blocks of 256.
// Each block reduces 128 pixels x 64 channels, then 64 f32 atomics.
// sums layout: [b][k'] with k' = t*64 + c  (de-interleaved order)
// ---------------------------------------------------------------------------
__global__ __launch_bounds__(256) void k1_gap(const float* __restrict__ t0,
                                              const float* __restrict__ t1,
                                              const float* __restrict__ t2,
                                              float* __restrict__ sums) {
    __shared__ float red[256];
    int bid = blockIdx.x;
    int chunk = bid & 127;
    int tsr = (bid >> 7) % 3;
    int b = bid / 384;
    const float* src = (tsr == 0 ? t0 : (tsr == 1 ? t1 : t2));
    src += (size_t)(b * PIX_PER_B + chunk * TILE_PIX) * CC;

    int tid = threadIdx.x;
    int c = tid & 63;
    int pr = tid >> 6;           // 0..3
    float acc = 0.0f;
    #pragma unroll 4
    for (int i = 0; i < 32; ++i) {
        acc += src[(size_t)(pr + 4 * i) * CC + c];   // coalesced: stride-256 f32
    }
    red[tid] = acc;
    __syncthreads();
    if (pr == 0) {
        float tot = red[c] + red[c + 64] + red[c + 128] + red[c + 192];
        atomicAdd(&sums[b * C3 + tsr * CC + c], tot);
    }
}

// ---------------------------------------------------------------------------
// Kernel 2: one block. SE MLP + build per-batch folded, transposed bf16
// weights: wt[b][f][k'] = sigmoid(...)[b, j] * w_conv[j, f], j = 3*(k'%64)+k'/64
// ---------------------------------------------------------------------------
__global__ __launch_bounds__(256) void k2_se_fold(const float* __restrict__ sums,
                                                  const float* __restrict__ w_se1,  // [192][12]
                                                  const float* __restrict__ w_se2,  // [12][192]
                                                  const float* __restrict__ w_conv, // [192][64]
                                                  unsigned short* __restrict__ wt) {
    __shared__ float m[BB * C3];      // means, interleaved-channel order j
    __shared__ float hvec[BB * HID];
    __shared__ float svec[BB * C3];
    int tid = threadIdx.x;

    const float inv_hw = 1.0f / (float)PIX_PER_B;
    for (int idx = tid; idx < BB * C3; idx += 256) {
        int b = idx / C3, j = idx % C3;
        int c = j / 3, t = j % 3;
        m[idx] = sums[b * C3 + t * CC + c] * inv_hw;
    }
    __syncthreads();

    if (tid < BB * HID) {
        int b = tid / HID, r = tid % HID;
        float acc = 0.0f;
        for (int j = 0; j < C3; ++j) acc += m[b * C3 + j] * w_se1[j * HID + r];
        hvec[tid] = fmaxf(acc, 0.0f);
    }
    __syncthreads();

    for (int idx = tid; idx < BB * C3; idx += 256) {
        int b = idx / C3, j = idx % C3;
        float acc = 0.0f;
        #pragma unroll
        for (int r = 0; r < HID; ++r) acc += hvec[b * HID + r] * w_se2[r * C3 + j];
        svec[idx] = 1.0f / (1.0f + __expf(-acc));
    }
    __syncthreads();

    for (int idx = tid; idx < WT_ELEMS; idx += 256) {
        int b = idx / WT_PER_B;
        int rem = idx % WT_PER_B;
        int f = rem / C3;
        int kp = rem % C3;
        int c = kp & 63, t = kp >> 6;
        int j = 3 * c + t;
        float v = svec[b * C3 + j] * w_conv[j * CC + f];
        wt[idx] = f2bf(v);
    }
}

// ---------------------------------------------------------------------------
// Kernel 3 (main): per-pixel 192->64 bf16 WMMA matmul + bias + ReLU.
// grid = 1024 blocks (128 per batch), 256 threads = 8 wave32.
// LDS: wlds[64][192] bf16 (24 KB) + xlds[128][192] bf16 (48 KB) = 72 KB dyn.
// Weight slab DMA'd by the Tensor Data Mover; X tile converted f32->bf16 by
// VALU and staged with ds_store_b64. Each wave: 16 pixels x 64 outs via
// four f32 accumulators, K=192 in 6 steps of v_wmma_f32_16x16x32_bf16.
// ---------------------------------------------------------------------------
__global__ __launch_bounds__(256) void k3_main(const float* __restrict__ t0,
                                               const float* __restrict__ t1,
                                               const float* __restrict__ t2,
                                               const unsigned short* __restrict__ wt,
                                               const float* __restrict__ bconv,
                                               float* __restrict__ out) {
    extern __shared__ __align__(16) unsigned char smem[];
    unsigned short* wlds = (unsigned short*)smem;                 // [64][192] @ LDS offset 0
    unsigned short* xlds = (unsigned short*)(smem + CC * C3 * 2); // [128][192]

    int tid = threadIdx.x;
    int b = blockIdx.x >> 7;               // 128 tiles per batch
    int tile = blockIdx.x & 127;
    int pbase = b * PIX_PER_B + tile * TILE_PIX;

    // --- stage per-batch folded weights (24 KB) into LDS -------------------
#if __has_builtin(__builtin_amdgcn_tensor_load_to_lds) && __has_builtin(__builtin_amdgcn_s_wait_tensorcnt)
    if (tid < 32) {   // one TDM issue per workgroup (wave 0; TDM ignores EXEC)
        unsigned long long ga =
            (unsigned long long)(const unsigned short*)(wt + (size_t)b * WT_PER_B);
        // D# group 0: count=1 (valid user desc), lds_addr=0 (dynamic LDS base),
        //             global_addr[56:0], type=2 ("image")
        u32x4 g0;
        g0[0] = 1u;
        g0[1] = 0u;                                   // lds_addr (bytes)
        g0[2] = (unsigned int)(ga & 0xFFFFFFFFull);   // global_addr[31:0]
        g0[3] = (unsigned int)((ga >> 32) & 0x1FFFFFFull) | (2u << 30); // [56:32] | type=2
        // D# group 1: data_size=2B; tensor = 12288 x 1; tile = 12288 x 1;
        //             tensor_dim0_stride = 12288
        i32x8 g1;
        g1[0] = (int)(1u << 16);          // data_size=1 (2 bytes); wg_mask=0
        g1[1] = (int)(12288u << 16);      // tensor_dim0[15:0] @ bits 63:48
        g1[2] = (int)(1u << 16);          // tensor_dim0[31:16]=0 | tensor_dim1=1
        g1[3] = (int)(12288u << 16);      // tensor_dim1[31:16]=0 | tile_dim0=12288
        g1[4] = 1;                        // tile_dim1=1, tile_dim2=0
        g1[5] = 12288;                    // tensor_dim0_stride[31:0]
        g1[6] = 0;                        // stride hi / tensor_dim1_stride lo
        g1[7] = 0;
        i32x4 g2 = {0, 0, 0, 0};
        i32x4 g3 = {0, 0, 0, 0};
#if defined(__clang_major__) && (__clang_major__ >= 23)
        i32x8 g4 = {0, 0, 0, 0, 0, 0, 0, 0};
        __builtin_amdgcn_tensor_load_to_lds(g0, g1, g2, g3, g4, 0);
#else
        __builtin_amdgcn_tensor_load_to_lds(g0, g1, g2, g3, 0);
#endif
        __builtin_amdgcn_s_wait_tensorcnt(0);
    }
#else
    {
        const uint4* wsrc = (const uint4*)(wt + (size_t)b * WT_PER_B);
        uint4* wdst = (uint4*)wlds;
        #pragma unroll
        for (int i = 0; i < 6; ++i) wdst[tid + i * 256] = wsrc[tid + i * 256];
    }
#endif

    // --- stage X tile: 3 tensors x 128 pix x 64 ch, f32 -> bf16 ------------
    {
        const float* tins[3] = { t0, t1, t2 };
        #pragma unroll
        for (int t = 0; t < 3; ++t) {
            const float4* src = (const float4*)(tins[t] + (size_t)pbase * CC);
            #pragma unroll
            for (int i = 0; i < 8; ++i) {
                int f4i = tid + i * 256;          // 0..2047
                float4 v = src[f4i];
                int pix = f4i >> 4;               // 64 ch = 16 float4 per pixel
                int c = (f4i & 15) * 4;
                US4 p;
                p.a = f2bf(v.x); p.b = f2bf(v.y); p.c = f2bf(v.z); p.d = f2bf(v.w);
                *(US4*)(xlds + pix * C3 + t * CC + c) = p;
            }
        }
    }
    __syncthreads();

    // --- WMMA compute ------------------------------------------------------
    int lane = tid & 31;
    int wave = tid >> 5;                   // 0..7 -> 16-pixel slab each
    int row = lane & 15;                   // M (pixels) / N (outs) per layout
    int hi = lane >> 4;                    // lane half selector

    v8f acc0 = {}, acc1 = {}, acc2 = {}, acc3 = {};

    const unsigned short* xrow = xlds + (wave * 16 + row) * C3;

    #pragma unroll
    for (int kk = 0; kk < 6; ++kk) {
        int k0 = kk * 32;
        // A fragment (16x32 bf16): elems 0..7 -> K = k0+8*hi+e,
        //                          elems 8..15 -> K = k0+16+8*hi+e
        V16U a;
        a.h[0] = *(const v8bf*)(xrow + k0 + 8 * hi);
        a.h[1] = *(const v8bf*)(xrow + k0 + 16 + 8 * hi);

        // B fragments (32x16 bf16): lane col = row, same K order as A
        #pragma unroll
        for (int n = 0; n < 4; ++n) {
            const unsigned short* wrow = wlds + (n * 16 + row) * C3 + k0;
            V16U bb;
            bb.h[0] = *(const v8bf*)(wrow + 8 * hi);
            bb.h[1] = *(const v8bf*)(wrow + 16 + 8 * hi);
            v8f* accp = (n == 0 ? &acc0 : n == 1 ? &acc1 : n == 2 ? &acc2 : &acc3);
            *accp = __builtin_amdgcn_wmma_f32_16x16x32_bf16(
                false, a.v, false, bb.v, (short)0, *accp, false, false);
        }
    }

    // --- bias + ReLU + store ----------------------------------------------
    // D layout: VGPR r -> lanes0-15: M=r, N=lane; lanes16-31: M=r+8, N=lane-16
    int pwave = pbase + wave * 16;
    #pragma unroll
    for (int n = 0; n < 4; ++n) {
        int f = n * 16 + row;
        float bias = bconv[f];
        const v8f& acc = (n == 0 ? acc0 : n == 1 ? acc1 : n == 2 ? acc2 : acc3);
        #pragma unroll
        for (int r = 0; r < 8; ++r) {
            int pix = pwave + r + 8 * hi;
            out[(size_t)pix * CC + f] = fmaxf(acc[r] + bias, 0.0f);
        }
    }
}

// ---------------------------------------------------------------------------
extern "C" void kernel_launch(void* const* d_in, const int* in_sizes, int n_in,
                              void* d_out, int out_size, void* d_ws, size_t ws_size,
                              hipStream_t stream) {
    const float* t1     = (const float*)d_in[0];
    const float* t2     = (const float*)d_in[1];
    const float* t3     = (const float*)d_in[2];
    const float* w_se1  = (const float*)d_in[3];
    const float* w_se2  = (const float*)d_in[4];
    const float* w_conv = (const float*)d_in[5];
    const float* b_conv = (const float*)d_in[6];
    float* out = (float*)d_out;

    float* ws_sums = (float*)d_ws;
    unsigned short* ws_wt = (unsigned short*)((char*)d_ws + WT_BYTE_OFF);

    k0_zero<<<(SUMS_ELEMS + 255) / 256, 256, 0, stream>>>(ws_sums);
    k1_gap<<<BB * 3 * (PIX_PER_B / TILE_PIX), 256, 0, stream>>>(t1, t2, t3, ws_sums);
    k2_se_fold<<<1, 256, 0, stream>>>(ws_sums, w_se1, w_se2, w_conv, ws_wt);

    size_t lds_bytes = (size_t)(CC * C3 + TILE_PIX * C3) * 2;   // 73728
    k3_main<<<NPIX / TILE_PIX, 256, lds_bytes, stream>>>(t1, t2, t3, ws_wt,
                                                         b_conv, out);
}
